// RNN_23441931501743
// MI455X (gfx1250) — compile-verified, matrix operand order checked by reading, not run on hardware
//
#include <hip/hip_runtime.h>
#include <hip/hip_bf16.h>
#include <math.h>

// Problem dims (fixed by the reference)
#define T_STEPS 1024
#define BATCH   64
#define IN_DIM  512
#define H_DIM   1024
#define NWG     64     // persistent workgroups in the scan (= H_DIM/16)

typedef __attribute__((ext_vector_type(16))) __bf16 v16bf;
typedef __attribute__((ext_vector_type(8)))  float  v8f;
typedef __attribute__((ext_vector_type(4)))  unsigned uv4;

union ABFrag { v16bf v; unsigned u[8]; uv4 q[2]; };

__device__ __forceinline__ unsigned short f2bf(float f) {
  unsigned u = __float_as_uint(f);
  unsigned r = u + 0x7FFFu + ((u >> 16) & 1u);   // round-to-nearest-even
  return (unsigned short)(r >> 16);
}
__device__ __forceinline__ unsigned pack_bf2(float lo, float hi) {
  return (unsigned)f2bf(lo) | ((unsigned)f2bf(hi) << 16);
}
__device__ __forceinline__ float fast_tanh(float v) {
#if __has_builtin(__builtin_amdgcn_tanhf)
  return __builtin_amdgcn_tanhf(v);     // gfx1250 v_tanh_f32
#else
  return tanhf(v);
#endif
}

// ---------------------------------------------------------------------------
// Prep: W (IN+H, H) f32 row-major -> WxT (H x IN) bf16, WhT (H x H) bf16
// (transposed so a WMMA fragment's K run is contiguous -> b128 loads)
// ---------------------------------------------------------------------------
__global__ void prep_weights(const float* __restrict__ W,
                             unsigned short* __restrict__ WxT,
                             unsigned short* __restrict__ WhT) {
  int idx = blockIdx.x * blockDim.x + threadIdx.x;
  if (idx >= (IN_DIM + H_DIM) * H_DIM) return;
  int k = idx / H_DIM, n = idx % H_DIM;
  unsigned short v = f2bf(W[idx]);
  if (k < IN_DIM) WxT[n * IN_DIM + k] = v;
  else            WhT[n * H_DIM + (k - IN_DIM)] = v;
}

// x (T*B*IN f32) -> packed bf16 (so GEMM A-fragments are plain b128 loads;
// 64 MB result is L2-resident for the 64 N-tile re-reads)
__global__ void prep_x(const float* __restrict__ x, unsigned* __restrict__ xbf) {
  int idx = blockIdx.x * blockDim.x + threadIdx.x;   // u32 output index
  if (idx >= T_STEPS * BATCH * IN_DIM / 2) return;
  float2 xv = ((const float2*)x)[idx];
  xbf[idx] = pack_bf2(xv.x, xv.y);
}

__global__ void prep_hidden(const float* __restrict__ h0,
                            unsigned short* __restrict__ hbuf0,
                            unsigned* __restrict__ counter) {
  int idx = blockIdx.x * blockDim.x + threadIdx.x;
  if (idx < BATCH * H_DIM) hbuf0[idx] = f2bf(h0[idx]);
  if (idx == 0) *counter = 0u;   // reset grid-barrier counter every launch
}

// ---------------------------------------------------------------------------
// xproj = x @ Wx + b  -> d_out  (M=T*B=65536, K=512, N=1024)
// block = 8 waves sharing one N-tile; WxT slab (16x512 bf16 = 16 KB) in LDS.
// Per WMMA: 2 global b128 (A) + 2 ds b128 (B) + 1 v_wmma.
// ---------------------------------------------------------------------------
__global__ void __launch_bounds__(256)
xproj_gemm(const unsigned* __restrict__ xbf_u32,
           const unsigned* __restrict__ WxT_u32,
           const float* __restrict__ bias,
           float* __restrict__ out) {
  __shared__ unsigned ldsB[16 * IN_DIM / 2];   // 16 KB

  const int lane = threadIdx.x & 31;
  const int wave = threadIdx.x >> 5;
  const int half = lane >> 4;
  const int l16  = lane & 15;
  const int mtile = blockIdx.x * 8 + wave;   // [0, 4096)
  const int ntile = blockIdx.y;              // [0, 64)
  const int row  = mtile * 16 + l16;         // A-fragment row (M)
  const int ncol = ntile * 16 + l16;         // B-fragment column (N)

  // stage this block's WxT slab (rows ntile*16..+15, contiguous) into LDS
  {
    const unsigned* src = WxT_u32 + (size_t)ntile * 16 * (IN_DIM / 2);
    for (int i = threadIdx.x; i < 16 * IN_DIM / 2; i += 256)
      ldsB[i] = src[i];
  }
  __syncthreads();

  const unsigned* arow = xbf_u32 + (size_t)row * (IN_DIM / 2);
  const unsigned* brow = &ldsB[l16 * (IN_DIM / 2)];

  v8f c = {};
  #pragma unroll 4
  for (int k0 = 0; k0 < IN_DIM; k0 += 32) {
    ABFrag a, b;
    const int kb = (k0 >> 1) + 4 * half;     // u32 index of first K pair
    a.q[0] = *(const uv4*)(arow + kb);
    a.q[1] = *(const uv4*)(arow + kb + 8);
    b.q[0] = *(const uv4*)(brow + kb);
    b.q[1] = *(const uv4*)(brow + kb + 8);
    c = __builtin_amdgcn_wmma_f32_16x16x32_bf16(false, a.v, false, b.v,
                                                (short)0, c, false, false);
  }

  const float bv = bias[ncol];
  #pragma unroll
  for (int r = 0; r < 8; ++r) {
    int m = mtile * 16 + r + 8 * half;
    out[(size_t)m * H_DIM + ncol] = c[r] + bv;
  }
}

// ---------------------------------------------------------------------------
// Persistent scan kernel: h = tanh(xproj_t + h @ Wh), 1024 sequential steps.
// 64 WGs x 128 threads (4 waves). WG owns 16 hidden columns; its 16x1024 bf16
// slab of WhT lives in LDS (32 KB). Wave w computes batch tile w (16 rows).
// h double-buffered in bf16 in global scratch; monotonic atomic grid barrier.
// ---------------------------------------------------------------------------
__global__ void __launch_bounds__(128)
rnn_scan(const unsigned* __restrict__ WhT_u32,
         unsigned short* __restrict__ hbuf0,
         unsigned short* __restrict__ hbuf1,
         unsigned* __restrict__ counter,
         float* __restrict__ out) {
  __shared__ unsigned ldsW[16 * H_DIM / 2];   // 32 KB: 16 columns x 1024 bf16

  const int lane = threadIdx.x & 31;
  const int wave = threadIdx.x >> 5;          // batch tile 0..3
  const int half = lane >> 4;
  const int l16  = lane & 15;
  const int wg   = blockIdx.x;                // n-tile 0..63

  // Stage this WG's Wh slab into LDS once (rows wg*16.. of WhT, contiguous)
  {
    const unsigned* src = WhT_u32 + (size_t)wg * 16 * (H_DIM / 2);
    for (int i = threadIdx.x; i < 16 * H_DIM / 2; i += 128)
      ldsW[i] = src[i];
  }
  __syncthreads();

  const int mrow = wave * 16 + l16;           // A-fragment batch row
  const int ncol = wg * 16 + l16;             // output hidden column
  const unsigned* brow = &ldsW[l16 * (H_DIM / 2)];

  for (int t = 0; t < T_STEPS; ++t) {
    const unsigned* hrd = (const unsigned*)((t & 1) ? hbuf1 : hbuf0);
    unsigned short* hwr = (t & 1) ? hbuf0 : hbuf1;
    const unsigned* arow = hrd + mrow * (H_DIM / 2);

    v8f c = {};
    #pragma unroll 4
    for (int ki = 0; ki < H_DIM / 32; ++ki) {
      const int kb = ki * 16 + 4 * half;      // u32 index of first K pair
      ABFrag a, b;
      a.q[0] = *(const uv4*)(arow + kb);
      a.q[1] = *(const uv4*)(arow + kb + 8);
      b.q[0] = *(const uv4*)(brow + kb);
      b.q[1] = *(const uv4*)(brow + kb + 8);
      c = __builtin_amdgcn_wmma_f32_16x16x32_bf16(false, a.v, false, b.v,
                                                  (short)0, c, false, false);
    }

    // h_next = tanh(xproj + h@Wh); xproj lives in d_out and is overwritten
    float* outp = out + (size_t)t * (BATCH * H_DIM);
    #pragma unroll
    for (int r = 0; r < 8; ++r) {
      int m = wave * 16 + r + 8 * half;
      size_t idx = (size_t)m * H_DIM + ncol;
      float hn = fast_tanh(c[r] + outp[idx]);
      outp[idx] = hn;
      hwr[idx]  = f2bf(hn);
    }

    // device-wide barrier: make step-t h visible before anyone reads it
    __threadfence();
    __syncthreads();
    if (threadIdx.x == 0) {
      __hip_atomic_fetch_add(counter, 1u, __ATOMIC_ACQ_REL,
                             __HIP_MEMORY_SCOPE_AGENT);
      const unsigned target = (unsigned)(t + 1) * NWG;
      while (__hip_atomic_load(counter, __ATOMIC_ACQUIRE,
                               __HIP_MEMORY_SCOPE_AGENT) < target)
        __builtin_amdgcn_s_sleep(1);
    }
    __syncthreads();
  }
}

// ---------------------------------------------------------------------------
extern "C" void kernel_launch(void* const* d_in, const int* in_sizes, int n_in,
                              void* d_out, int out_size, void* d_ws, size_t ws_size,
                              hipStream_t stream) {
  (void)in_sizes; (void)n_in; (void)out_size; (void)ws_size;
  const float* x = (const float*)d_in[0];   // (T, B, IN)
  const float* h = (const float*)d_in[1];   // (B, H)
  const float* W = (const float*)d_in[2];   // (IN+H, H)
  const float* b = (const float*)d_in[3];   // (H,)
  float* out = (float*)d_out;               // (T, B, H)

  char* ws = (char*)d_ws;
  unsigned short* WxT = (unsigned short*)ws;                              // 1 MB
  unsigned short* WhT = (unsigned short*)(ws + (1u << 20));               // 2 MB
  unsigned short* hb0 = (unsigned short*)(ws + 3u * (1u << 20));          // 128 KB
  unsigned short* hb1 = (unsigned short*)(ws + 3u * (1u << 20) + (128u << 10));
  unsigned*   counter = (unsigned*)(ws + 3u * (1u << 20) + (256u << 10));
  unsigned*   xbf     = (unsigned*)(ws + 4u * (1u << 20));                // 64 MB

  prep_weights<<<((IN_DIM + H_DIM) * H_DIM + 255) / 256, 256, 0, stream>>>(W, WxT, WhT);
  prep_x<<<(T_STEPS * BATCH * IN_DIM / 2 + 255) / 256, 256, 0, stream>>>(x, xbf);
  prep_hidden<<<(BATCH * H_DIM + 255) / 256, 256, 0, stream>>>(h, hb0, counter);

  dim3 g1((T_STEPS * BATCH / 16) / 8, H_DIM / 16);   // (512, 64)
  xproj_gemm<<<g1, 256, 0, stream>>>(xbf, (const unsigned*)WxT, b, out);

  rnn_scan<<<NWG, 128, 0, stream>>>((const unsigned*)WhT, hb0, hb1, counter, out);
}
